// HTSAD_25451976196840
// MI455X (gfx1250) — matrix operand order repaired and checked by reading, not compiled, not run on hardware
//
#include <hip/hip_runtime.h>
#include <hip/hip_bf16.h>
#include <math.h>

// ---------------------------------------------------------------------------
// Types for CDNA5 WMMA (wave32, V_WMMA_F32_16X16X32_BF16)
// ---------------------------------------------------------------------------
typedef __bf16 bf16;
typedef __attribute__((ext_vector_type(16))) __bf16 v16bf;
typedef __attribute__((ext_vector_type(8)))  __bf16 bf16x8;
typedef __attribute__((ext_vector_type(8)))  float  v8f;

static __device__ inline v8f v8f_zero() {
    v8f z;
#pragma unroll
    for (int i = 0; i < 8; ++i) z[i] = 0.f;
    return z;
}

// Assemble a 16-element bf16 fragment from two contiguous 8-element runs.
static __device__ inline v16bf load_frag2(const bf16* p0, const bf16* p1) {
    bf16x8 lo = *(const bf16x8*)p0;
    bf16x8 hi = *(const bf16x8*)p1;
    v16bf r;
#pragma unroll
    for (int i = 0; i < 8; ++i) { r[i] = lo[i]; r[i + 8] = hi[i]; }
    return r;
}

// Gather 16 strided f32 values (one weight column) and round to bf16.
static __device__ inline v16bf gather_bf16(const float* base, int stride) {
    v16bf r;
#pragma unroll
    for (int e = 0; e < 16; ++e) r[e] = (bf16)base[(size_t)e * stride];
    return r;
}

static __device__ inline v8f wmma_bf16(v16bf a, v16bf b, v8f c) {
    return __builtin_amdgcn_wmma_f32_16x16x32_bf16(
        /*neg_a=*/false, a, /*neg_b=*/false, b,
        /*c_mod=*/(short)0, c, /*reuse_a=*/false, /*reuse_b=*/false);
}

static __device__ inline float sigm(float x) { return 1.f / (1.f + __expf(-x)); }

#define B_  64
#define S_  4096
#define EMB_ 128
#define HS_  256
#define G_   1024   // 4*HS

// ---------------------------------------------------------------------------
// Block decomposition shared by prep kernels and the scan:
// block blk = bg*1024 + tb ;  local row lr = mt*16 + m  <->
//   batch b = bg*16 + m ,  timestep t = tb*4 + mt.
// An M-tile is 16 batches at one timestep == the scan's C tile, so prep
// epilogues write single b128 stores in the scan's swizzled layout:
//   xw element (t, bg, wave w, lane l, gate-tile nt, e) at
//       ((t*4+bg)*512 + w*32 + l)*32 + nt*8 + e          (bf16)
//   j  element at ((t*4+bg)*512 + w*32 + l)*8 + e        (bf16)
// so each scan lane reads ONE contiguous 64B chunk (xw) + 16B (j) per step.
// ---------------------------------------------------------------------------

// ---------------------------------------------------------------------------
// Prep kernel 1: s_emb = event@Ve ; x = s_emb + 2*(vc@Vc + tanh(vn@Vn))
//   xw = swizzle(x@Wx + bias) ; se[blk][lr][:] = s_emb (block-local layout)
// 64 rows per workgroup, 256 threads (8 waves). x@Wx via WMMA bf16.
// ---------------------------------------------------------------------------
__global__ __launch_bounds__(256) void prep_embed(
    const float* __restrict__ event, const float* __restrict__ vcp,
    const float* __restrict__ vnp,  const float* __restrict__ Ve,
    const float* __restrict__ Vc,   const float* __restrict__ Vn,
    const float* __restrict__ Wx,   const float* __restrict__ bias,
    bf16* __restrict__ xw, bf16* __restrict__ se)
{
    __shared__ float ev_s[64 * 64];
    __shared__ float vc_s[64 * 32];
    __shared__ float vn_s[64 * 16];
    __shared__ bf16  x_s[64 * 136];     // padded stride: conflict-free frags

    const int tid = threadIdx.x;
    const int blk = blockIdx.x;
    const int bg  = blk >> 10;          // batch group 0..3
    const int tb  = blk & 1023;         // time block (4 consecutive steps)

    for (int i = tid; i < 64 * 64; i += 256) {
        const int lr = i >> 6, e = i & 63;
        const size_t r = (size_t)(bg * 16 + (lr & 15)) * S_ + (tb * 4 + (lr >> 4));
        ev_s[i] = event[r * 64 + e];
    }
    for (int i = tid; i < 64 * 32; i += 256) {
        const int lr = i >> 5, c = i & 31;
        const size_t r = (size_t)(bg * 16 + (lr & 15)) * S_ + (tb * 4 + (lr >> 4));
        vc_s[i] = vcp[r * 32 + c];
    }
    for (int i = tid; i < 64 * 16; i += 256) {
        const int lr = i >> 4, q = i & 15;
        const size_t r = (size_t)(bg * 16 + (lr & 15)) * S_ + (tb * 4 + (lr >> 4));
        vn_s[i] = vnp[r * 16 + q];
    }
    __syncthreads();

    {   // VALU phase: small-K projections (K = 64/32/16), 32 outputs/thread
        const int col = tid & 127;
        const int r0  = (tid >> 7) * 32;
        for (int rr = 0; rr < 32; ++rr) {
            const int lr = r0 + rr;
            float s = 0.f, cc = 0.f, nn = 0.f;
#pragma unroll 4
            for (int e = 0; e < 64; ++e) s  += ev_s[lr * 64 + e] * Ve[e * 128 + col];
#pragma unroll 4
            for (int c = 0; c < 32; ++c) cc += vc_s[lr * 32 + c] * Vc[c * 128 + col];
#pragma unroll 4
            for (int q = 0; q < 16; ++q) nn += vn_s[lr * 16 + q] * Vn[q * 128 + col];
            const float x = s + 2.f * (cc + tanhf(nn));
            x_s[lr * 136 + col] = (bf16)x;
            se[(size_t)blk * (64 * 128) + lr * 128 + col] = (bf16)s;
        }
    }
    __syncthreads();

    // WMMA phase: [64,128] @ Wx[128,1024]; wave w owns 128 output cols.
    const int lane = tid & 31, wave = tid >> 5;
    const bool hi = lane >= 16;
    const int ln  = lane & 15;
    const int off = hi ? 16 : 0;

    v8f acc[4][8];
#pragma unroll
    for (int mt = 0; mt < 4; ++mt)
#pragma unroll
        for (int nt = 0; nt < 8; ++nt) acc[mt][nt] = v8f_zero();

#pragma unroll
    for (int kt = 0; kt < 4; ++kt) {
        v16bf A[4];
#pragma unroll
        for (int mt = 0; mt < 4; ++mt) {
            const bf16* rp = &x_s[(mt * 16 + ln) * 136 + kt * 32 + off];
            A[mt] = load_frag2(rp, rp + 8);
        }
        const int kb = kt * 32 + (hi ? 16 : 0);
#pragma unroll
        for (int nt = 0; nt < 8; ++nt) {
            const int col = wave * 128 + nt * 16 + ln;
            v16bf Bf = gather_bf16(&Wx[(size_t)kb * G_ + col], G_);
#pragma unroll
            for (int mt = 0; mt < 4; ++mt) acc[mt][nt] = wmma_bf16(A[mt], Bf, acc[mt][nt]);
        }
    }
    // Epilogue: one b128 store per (mt,nt) in scan-swizzled layout
#pragma unroll
    for (int nt = 0; nt < 8; ++nt) {
        const int col = wave * 128 + nt * 16 + ln;
        const float bb = bias[col];
        const int wq  = wave * 8 + nt;        // global 16-col tile id 0..63
        const int ntp = wq >> 4;              // scan gate tile (i/f/g/o)
        const int wp  = wq & 15;              // scan wave owning these cols
#pragma unroll
        for (int mt = 0; mt < 4; ++mt) {
            const int t = tb * 4 + mt;
            bf16x8 pk;
#pragma unroll
            for (int e = 0; e < 8; ++e) pk[e] = (bf16)(acc[mt][nt][e] + bb);
            const size_t base = ((size_t)t * 4 + bg) * 512 + wp * 32 + lane;
            *(bf16x8*)(xw + base * 32 + ntp * 8) = pk;
        }
    }
}

// ---------------------------------------------------------------------------
// Prep kernel 2: j = sigmoid( tanh(s_emb@Wef1 + bef1) @ Wef3 + bef3 )
// Two chained WMMA GEMMs (K=128 each); j stored in scan-swizzled layout.
// ---------------------------------------------------------------------------
__global__ __launch_bounds__(256) void prep_gate(
    const bf16* __restrict__ se, const float* __restrict__ Wef1,
    const float* __restrict__ bef1, const float* __restrict__ Wef3,
    const float* __restrict__ bef3, bf16* __restrict__ jg)
{
    __shared__ bf16 a_s[64 * 136];
    __shared__ bf16 y_s[64 * 136];
    const int tid = threadIdx.x;
    const int blk = blockIdx.x;
    const int bg  = blk >> 10;
    const int tb  = blk & 1023;

    for (int i = tid; i < 64 * 128; i += 256)
        a_s[(i >> 7) * 136 + (i & 127)] = se[(size_t)blk * (64 * 128) + i];
    __syncthreads();

    const int lane = tid & 31, wave = tid >> 5;
    const bool hi = lane >= 16;
    const int ln  = lane & 15;
    const int off = hi ? 16 : 0;

    {   // GEMM1: N=128, wave owns one 16-col tile
        const int col = wave * 16 + ln;
        v8f acc[4];
#pragma unroll
        for (int mt = 0; mt < 4; ++mt) acc[mt] = v8f_zero();
#pragma unroll
        for (int kt = 0; kt < 4; ++kt) {
            const int kb = kt * 32 + (hi ? 16 : 0);
            v16bf Bf = gather_bf16(&Wef1[(size_t)kb * 128 + col], 128);
#pragma unroll
            for (int mt = 0; mt < 4; ++mt) {
                const bf16* rp = &a_s[(mt * 16 + ln) * 136 + kt * 32 + off];
                acc[mt] = wmma_bf16(load_frag2(rp, rp + 8), Bf, acc[mt]);
            }
        }
        const float b1 = bef1[col];
#pragma unroll
        for (int mt = 0; mt < 4; ++mt)
#pragma unroll
            for (int e = 0; e < 8; ++e) {
                const int m = e + (hi ? 8 : 0);
                y_s[(mt * 16 + m) * 136 + col] = (bf16)tanhf(acc[mt][e] + b1);
            }
    }
    __syncthreads();

    {   // GEMM2: N=256, wave owns two 16-col tiles
        v8f acc[4][2];
#pragma unroll
        for (int mt = 0; mt < 4; ++mt) { acc[mt][0] = v8f_zero(); acc[mt][1] = v8f_zero(); }
#pragma unroll
        for (int kt = 0; kt < 4; ++kt) {
            const int kb = kt * 32 + (hi ? 16 : 0);
            v16bf A[4];
#pragma unroll
            for (int mt = 0; mt < 4; ++mt) {
                const bf16* rp = &y_s[(mt * 16 + ln) * 136 + kt * 32 + off];
                A[mt] = load_frag2(rp, rp + 8);
            }
#pragma unroll
            for (int nt = 0; nt < 2; ++nt) {
                const int col = wave * 32 + nt * 16 + ln;
                v16bf Bf = gather_bf16(&Wef3[(size_t)kb * HS_ + col], HS_);
#pragma unroll
                for (int mt = 0; mt < 4; ++mt) acc[mt][nt] = wmma_bf16(A[mt], Bf, acc[mt][nt]);
            }
        }
        // Epilogue: one b128 store per (mt,nt) in scan-swizzled layout
#pragma unroll
        for (int nt = 0; nt < 2; ++nt) {
            const int col = wave * 32 + nt * 16 + ln;
            const float b3 = bef3[col];
            const int wq  = wave * 2 + nt;    // scan wave owning these cols
#pragma unroll
            for (int mt = 0; mt < 4; ++mt) {
                const int t = tb * 4 + mt;
                bf16x8 pk;
#pragma unroll
                for (int e = 0; e < 8; ++e) pk[e] = (bf16)sigm(acc[mt][nt][e] + b3);
                const size_t base = ((size_t)t * 4 + bg) * 512 + wq * 32 + lane;
                *(bf16x8*)(jg + base * 8) = pk;
            }
        }
    }
}

// ---------------------------------------------------------------------------
// Scan kernel: 4 workgroups x 512 threads (16 waves). WG bg owns batches
// [16bg,16bg+16). Wave w owns hidden slice n in [16w,16w+16) and gate columns
// {n, 256+n, 512+n, 768+n} -> i/f/g/o land in the same lane's accumulators.
// Wh split: K=0..127 bf16 in LDS (transposed, padded rows of 136),
// K=128..255 register resident. h,c carries are f32 registers; the bf16 copy
// of h (next step's WMMA A operand) ping-pongs between two LDS buffers so a
// single barrier per step suffices. xw/j streams are software-pipelined:
// step t+1's 64B+16B chunks load while step t's 32 WMMAs execute.
// ---------------------------------------------------------------------------
#define SCAN_LDS_BYTES (1024 * 136 * 2 + 2 * 16 * 264 * 2)   // 295,424 B

__global__ __launch_bounds__(512) void scan_kernel(
    const bf16* __restrict__ xw, const bf16* __restrict__ jg,
    const float* __restrict__ Wh, const float* __restrict__ Wc,
    const float* __restrict__ h0, const float* __restrict__ c0,
    const float* __restrict__ Wlin, const float* __restrict__ blin,
    float* __restrict__ out)
{
    extern __shared__ __align__(16) char smem[];
    bf16* WhL   = (bf16*)smem;                        // [1024][136]: WhL[col*136+k], k<128
    bf16* hbuf0 = (bf16*)(smem + 1024 * 136 * 2);     // [16][264] ping
    bf16* hbuf1 = hbuf0 + 16 * 264;                   // [16][264] pong

    const int tid  = threadIdx.x;
    const int lane = tid & 31, wave = tid >> 5;
    const bool hi  = lane >= 16;
    const int ln   = lane & 15;
    const int off  = hi ? 16 : 0;
    const int bg   = blockIdx.x;            // batch group
    const int bo   = bg * 16;               // batch offset
    const int n    = wave * 16 + ln;        // this lane's hidden column

    // Stage low-K half of Wh (transposed, bf16)
    for (int i = tid; i < 128 * 1024; i += 512) {
        const int k = i & 127, col = i >> 7;
        WhL[col * 136 + k] = (bf16)Wh[(size_t)k * G_ + col];
    }
    // Init ping buffer from h0
    for (int i = tid; i < 16 * 256; i += 512) {
        const int m = i >> 8, c = i & 255;
        hbuf0[m * 264 + c] = (bf16)h0[(size_t)(bo + m) * HS_ + c];
    }

    // High-K half of Wh, register resident (4 ktiles x 4 gate tiles)
    v16bf BH[4][4];
#pragma unroll
    for (int kt = 0; kt < 4; ++kt) {
        const int kb = 128 + kt * 32 + (hi ? 16 : 0);
#pragma unroll
        for (int nt = 0; nt < 4; ++nt)
            BH[kt][nt] = gather_bf16(&Wh[(size_t)kb * G_ + (nt * HS_ + n)], G_);
    }

    // Per-lane f32 carry state (C-tile layout: m = e + 8*hi)
    v8f cC, hC;
#pragma unroll
    for (int e = 0; e < 8; ++e) {
        const int m = e + (hi ? 8 : 0);
        cC[e] = c0[(size_t)(bo + m) * HS_ + n];
        hC[e] = h0[(size_t)(bo + m) * HS_ + n];
    }
    const float wc0 = Wc[n], wc1 = Wc[HS_ + n], wc2 = Wc[2 * HS_ + n];

    // Prefetch step 0's xw (64B) + j (16B) chunks
    bf16x8 xb[4], jb;
    {
        const size_t base = (size_t)bg * 512 + wave * 32 + lane;
        const bf16x8* xp = (const bf16x8*)(xw + base * 32);
        xb[0] = xp[0]; xb[1] = xp[1]; xb[2] = xp[2]; xb[3] = xp[3];
        jb = *(const bf16x8*)(jg + base * 8);
    }
    __syncthreads();

    for (int t = 0; t < S_; ++t) {
        const bf16* hr = (t & 1) ? hbuf1 : hbuf0;   // holds h_t
        bf16*       hw = (t & 1) ? hbuf0 : hbuf1;   // receives h_{t+1}

        // Seed accumulators with xw_t (bias folded in), grab j_t
        v8f acc[4];
#pragma unroll
        for (int nt = 0; nt < 4; ++nt)
#pragma unroll
            for (int e = 0; e < 8; ++e) acc[nt][e] = (float)xb[nt][e];
        float jv[8];
#pragma unroll
        for (int e = 0; e < 8; ++e) jv[e] = (float)jb[e];

        // Software-pipelined prefetch of step t+1's chunks
        if (t + 1 < S_) {
            const size_t base = ((size_t)(t + 1) * 4 + bg) * 512 + wave * 32 + lane;
            const bf16x8* xp = (const bf16x8*)(xw + base * 32);
            xb[0] = xp[0]; xb[1] = xp[1]; xb[2] = xp[2]; xb[3] = xp[3];
            jb = *(const bf16x8*)(jg + base * 8);
        }

        // gates += h @ Wh : 8 ktiles x 4 gate tiles = 32 WMMAs per wave
#pragma unroll
        for (int kt = 0; kt < 8; ++kt) {
            const bf16* ap = &hr[ln * 264 + kt * 32 + off];
            const v16bf A = load_frag2(ap, ap + 8);
            if (kt < 4) {
                const int kb = kt * 32 + (hi ? 16 : 0);
#pragma unroll
                for (int nt = 0; nt < 4; ++nt) {
                    const bf16* bp = &WhL[(nt * HS_ + n) * 136 + kb];
                    acc[nt] = wmma_bf16(A, load_frag2(bp, bp + 8), acc[nt]);
                }
            } else {
#pragma unroll
                for (int nt = 0; nt < 4; ++nt)
                    acc[nt] = wmma_bf16(A, BH[kt - 4][nt], acc[nt]);
            }
        }

        // Elementwise LSTM update, entirely in registers; write h_{t+1} bf16
#pragma unroll
        for (int e = 0; e < 8; ++e) {
            const int m = e + (hi ? 8 : 0);
            const float c = cC[e], h = hC[e], jt = jv[e];
            const float ig = sigm(acc[0][e] + c * wc0);
            const float fg = sigm(acc[1][e] + c * wc1);
            const float gg = tanhf(acc[2][e]);
            const float og = sigm(acc[3][e] + c * wc2);
            const float ch = fg * c + ig * gg;
            const float cn = jt * ch + (1.f - jt) * c;
            const float hn = jt * (og * tanhf(ch)) + (1.f - jt) * h;
            cC[e] = cn; hC[e] = hn;
            hw[m * 264 + n] = (bf16)hn;
        }
        __syncthreads();   // single barrier: hw complete, hr free for reuse
    }

    // Epilogue: out = h_T @ Wlin + blin. S_ even -> h_T lives in hbuf0.
    if (wave < 4) {
        const int col = wave * 16 + ln;
        v8f accO = v8f_zero();
#pragma unroll
        for (int kt = 0; kt < 8; ++kt) {
            const bf16* ap = &hbuf0[ln * 264 + kt * 32 + off];
            const int kb = kt * 32 + (hi ? 16 : 0);
            accO = wmma_bf16(load_frag2(ap, ap + 8),
                             gather_bf16(&Wlin[(size_t)kb * 64 + col], 64), accO);
        }
        const float bb = blin[col];
#pragma unroll
        for (int e = 0; e < 8; ++e) {
            const int m = e + (hi ? 8 : 0);
            out[(size_t)(bo + m) * 64 + col] = accO[e] + bb;
        }
    }
}

// ---------------------------------------------------------------------------
// Host launcher. Workspace: xw (512MB) + j (128MB) + s_emb (64MB), all bf16.
// ---------------------------------------------------------------------------
extern "C" void kernel_launch(void* const* d_in, const int* in_sizes, int n_in,
                              void* d_out, int out_size, void* d_ws, size_t ws_size,
                              hipStream_t stream) {
    const float* event = (const float*)d_in[0];
    // d_in[1] = time : unused by the reference
    const float* vcp  = (const float*)d_in[2];
    const float* vnp  = (const float*)d_in[3];
    const float* h0   = (const float*)d_in[4];
    const float* c0   = (const float*)d_in[5];
    const float* Wx   = (const float*)d_in[6];
    const float* Wh   = (const float*)d_in[7];
    const float* Wc   = (const float*)d_in[8];
    const float* bias = (const float*)d_in[9];
    const float* Ve   = (const float*)d_in[10];
    const float* Vc   = (const float*)d_in[11];
    const float* Vn   = (const float*)d_in[12];
    const float* Wlin = (const float*)d_in[13];
    const float* blin = (const float*)d_in[14];
    const float* Wef1 = (const float*)d_in[15];
    const float* bef1 = (const float*)d_in[16];
    const float* Wef3 = (const float*)d_in[17];
    const float* bef3 = (const float*)d_in[18];

    const size_t rows = (size_t)B_ * S_;
    bf16* xw = (bf16*)d_ws;                  // swizzled [S][4][512][32]
    bf16* jg = xw + rows * G_;               // swizzled [S][4][512][8]
    bf16* se = jg + rows * HS_;              // block-local [4096][64][128]

    prep_embed<<<rows / 64, 256, 0, stream>>>(event, vcp, vnp, Ve, Vc, Vn, Wx, bias, xw, se);
    prep_gate<<<rows / 64, 256, 0, stream>>>(se, Wef1, bef1, Wef3, bef3, jg);

    (void)hipFuncSetAttribute((const void*)scan_kernel,
                              hipFuncAttributeMaxDynamicSharedMemorySize,
                              SCAN_LDS_BYTES);
    scan_kernel<<<4, 512, SCAN_LDS_BYTES, stream>>>(xw, jg, Wh, Wc, h0, c0,
                                                    Wlin, blin, (float*)d_out);
}
